// ZImageAttention_6588479832562
// MI455X (gfx1250) — matrix operand order repaired and checked by reading, not compiled, not run on hardware
//
#include <hip/hip_runtime.h>
#include <hip/hip_bf16.h>

// ---------------------------------------------------------------------------
// FAVOR+ attention pipeline for MI455X (gfx1250).
// bf16 WMMA GEMMs with async-copy (global->LDS) double-buffered B tiles.
// B=2, S=4096, DIM=2048, H=16, HK=8, HD=128, M=128
// ---------------------------------------------------------------------------

#define BB   2
#define SS   4096
#define DIMD 2048
#define HH   16
#define HKK  8
#define HDD  128
#define MM   128
#define NTOK (BB * SS)          // 8192 tokens

typedef __bf16 bf16;
typedef __attribute__((ext_vector_type(16))) __bf16 v16bf;
typedef __attribute__((ext_vector_type(8)))  __bf16 v8bf;
typedef __attribute__((ext_vector_type(8)))  float  v8f;

// 128^-0.25 and 128^-0.5
#define CSCALE 0.29730177875068026f
#define MINV   0.08838834764831845f
#define EPS_F  1e-6f
#define EPS_R  1e-5f

// ------------------------- fragment helpers --------------------------------
__device__ __forceinline__ v16bf ld16(const bf16* p0, const bf16* p1) {
  v8bf a = *(const v8bf*)p0;
  v8bf b = *(const v8bf*)p1;
  v16bf r;
#pragma unroll
  for (int i = 0; i < 8; ++i) { r[i] = a[i]; r[i + 8] = b[i]; }
  return r;
}

// A row-major [row][K], 16x32 fragment at (row, k0).
// lane<16: M=lane, K=k0+{0..7,16..23}; lane>=16: M=lane-16, K=k0+{8..15,24..31}
__device__ __forceinline__ v16bf load_afrag(const bf16* A, size_t lda, int row,
                                            int k0, int lane) {
  int r = lane & 15, hi = lane >> 4;
  const bf16* p = A + (size_t)(row + r) * lda + k0 + hi * 8;
  return ld16(p, p + 16);
}

// CDNA5 async copy: 16 bytes global -> LDS per lane, tracked by ASYNCcnt.
__device__ __forceinline__ void async_cp16(unsigned lds_off, const bf16* g) {
  asm volatile("global_load_async_to_lds_b128 %0, %1, off"
               :: "v"(lds_off), "v"(g)
               : "memory");
}
__device__ __forceinline__ void wait_async0() {
  asm volatile("s_wait_asynccnt 0x0" ::: "memory");
}

// One 32-K step: pull all 8 B fragments from LDS into distinct registers
// (single ds clause, partial dscnt waits), then issue 8 WMMAs.
__device__ __forceinline__ void step_compute(const bf16* bbase, int n_, int hi,
                                             v16bf af, v8f acc[8]) {
  v16bf bfr[8];
#pragma unroll
  for (int t = 0; t < 8; ++t) {
    const bf16* bp = bbase + (16 * t + n_) * 32 + hi * 8;
    bfr[t] = ld16(bp, bp + 16);
  }
#pragma unroll
  for (int t = 0; t < 8; ++t)
    acc[t] = __builtin_amdgcn_wmma_f32_16x16x32_bf16(
        false, af, false, bfr[t], (short)0, acc[t], false, false);
}

// ---------------------------------------------------------------------------
// One wave computes 16 rows x 128 cols (8 C tiles).  B (stored [N][K], ldb)
// is staged through LDS with double-buffered async copies: all 4 waves of the
// block share the same 128x32 B tile.  Bs must hold 2*128*32 bf16 (16 KB).
// blockDim.x == 128.  K must be a multiple of 64 (even step count).
// ---------------------------------------------------------------------------
__device__ __forceinline__ void gemm_rowblock_lds(
    const bf16* __restrict__ A, size_t lda,
    const bf16* __restrict__ Bg, size_t ldb,
    bf16* Bs, int row, int K, v8f acc[8]) {
  const int tid  = threadIdx.x;          // 0..127: producer row = tid
  const int lane = tid & 31;
  const int n_ = lane & 15, hi = lane >> 4;

  const bf16* grow = Bg + (size_t)tid * ldb;

  // prime buffer 0 (rows 0..127, k = 0..31; 4 x b128 per thread)
#pragma unroll
  for (int i = 0; i < 4; ++i)
    async_cp16((unsigned)(size_t)(Bs + tid * 32 + i * 8), grow + i * 8);

  v16bf afA = load_afrag(A, lda, row, 0, lane);
  v16bf afB = afA;

  const int nsteps = K / 32;             // even for all users
#pragma unroll 2
  for (int stp = 0; stp < nsteps; ++stp) {
    // our async writes for the current buffer are done; sync the block so
    // every wave's slice of the tile is visible
    wait_async0();
    __syncthreads();

    // kick off DMA for the next k-chunk into the other buffer
    if (stp + 1 < nsteps) {
      const bf16* gnext = grow + (size_t)(stp + 1) * 32;
      bf16* dst = Bs + ((stp + 1) & 1) * (128 * 32) + tid * 32;
#pragma unroll
      for (int i = 0; i < 4; ++i)
        async_cp16((unsigned)(size_t)(dst + i * 8), gnext + i * 8);
      // prefetch next A fragment into the other ping-pong register
      if (stp & 1) afA = load_afrag(A, lda, row, (stp + 1) * 32, lane);
      else         afB = load_afrag(A, lda, row, (stp + 1) * 32, lane);
    }

    const bf16* bbase = Bs + (stp & 1) * (128 * 32);
    step_compute(bbase, n_, hi, (stp & 1) ? afB : afA, acc);
    __syncthreads();   // all reads of this buffer done before it is re-filled
  }
}

// reductions within the 16-lane half-group that shares a C row
__device__ __forceinline__ float hsum16(float v) {
  v += __shfl_xor(v, 1, 32);
  v += __shfl_xor(v, 2, 32);
  v += __shfl_xor(v, 4, 32);
  v += __shfl_xor(v, 8, 32);
  return v;
}
__device__ __forceinline__ float hmax16(float v) {
  v = fmaxf(v, __shfl_xor(v, 1, 32));
  v = fmaxf(v, __shfl_xor(v, 2, 32));
  v = fmaxf(v, __shfl_xor(v, 4, 32));
  v = fmaxf(v, __shfl_xor(v, 8, 32));
  return v;
}

// ------------------------- kernels -----------------------------------------

__global__ void k_cvt(const float* __restrict__ s, bf16* __restrict__ d, int n) {
  int i = blockIdx.x * blockDim.x + threadIdx.x;
  int str = gridDim.x * blockDim.x;
  for (; i < n; i += str) d[i] = (bf16)s[i];
}

// Q/K projection GEMM + RMSNorm + RoPE, bf16 output. For K also store the
// post-rope sum-of-squares.  grid = (NTOK/64, nHeads), block 128.
__global__ void __launch_bounds__(128)
k_qk_proj(const bf16* __restrict__ X, const bf16* __restrict__ W,
          const float* __restrict__ g, const float* __restrict__ fc,
          bf16* __restrict__ out, float* __restrict__ ssk, int ldo) {
  __shared__ bf16 Bs[2 * 128 * 32];
  int lane = threadIdx.x & 31;
  int wave = threadIdx.x >> 5;
  int row  = (blockIdx.x * 4 + wave) * 16;
  int h    = blockIdx.y;
  int n = lane & 15, hi = lane >> 4;

  v8f acc[8] = {};
  gemm_rowblock_lds(X, DIMD, W + (size_t)h * HDD * DIMD, DIMD, Bs, row, DIMD, acc);

  float gv[8];
#pragma unroll
  for (int t = 0; t < 8; ++t) gv[t] = g[16 * t + n];

#pragma unroll
  for (int j = 0; j < 8; ++j) {
    int tok = row + j + 8 * hi;
    float ss = 0.f;
#pragma unroll
    for (int t = 0; t < 8; ++t) { float v = acc[t][j]; ss += v * v; }
    ss = hsum16(ss);
    float scale = rsqrtf(ss * (1.f / (float)HDD) + EPS_R);

    float ov[8];
    float ss2 = 0.f;
#pragma unroll
    for (int t = 0; t < 8; ++t) {
      int c = 16 * t + n;
      float v = acc[t][j] * scale * gv[t];
      float other = __shfl_xor(v, 1, 32);
      size_t fbase = ((size_t)tok * (HDD / 2) + (c >> 1)) * 2;
      float cs = fc[fbase + 0];
      float sn = fc[fbase + 1];
      float o;
      if (n & 1) o = other * sn + v * cs;   // out[2i+1] = a*sin + b*cos
      else       o = v * cs - other * sn;   // out[2i]   = a*cos - b*sin
      ov[t] = o;
      ss2 += o * o;
    }
#pragma unroll
    for (int t = 0; t < 8; ++t)
      out[(size_t)tok * ldo + h * HDD + 16 * t + n] = (bf16)ov[t];
    if (ssk) {
      ss2 = hsum16(ss2);
      if (n == 0) ssk[(size_t)tok * HKK + h] = ss2;
    }
  }
}

// V projection GEMM, bf16 transposed store vT[b][hk][d][s]. grid=(NTOK/64, HK)
__global__ void __launch_bounds__(128)
k_v_proj(const bf16* __restrict__ X, const bf16* __restrict__ W,
         bf16* __restrict__ vT) {
  __shared__ bf16 Bs[2 * 128 * 32];
  int lane = threadIdx.x & 31;
  int wave = threadIdx.x >> 5;
  int row  = (blockIdx.x * 4 + wave) * 16;
  int h    = blockIdx.y;
  int n = lane & 15, hi = lane >> 4;

  v8f acc[8] = {};
  gemm_rowblock_lds(X, DIMD, W + (size_t)h * HDD * DIMD, DIMD, Bs, row, DIMD, acc);

#pragma unroll
  for (int j = 0; j < 8; ++j) {
    int tok = row + j + 8 * hi;
    int b = tok >> 12, s = tok & (SS - 1);
#pragma unroll
    for (int t = 0; t < 8; ++t) {
      int d = 16 * t + n;
      vT[(((size_t)(b * HKK + h) * HDD + d) * SS) + s] = (bf16)acc[t][j];
    }
  }
}

// FAVOR q: projection GEMM + row max + exp -> bf16 q_phi[tok][h*128+m]
__global__ void __launch_bounds__(128)
k_favor_q(const bf16* __restrict__ Q, const bf16* __restrict__ P,
          bf16* __restrict__ qphi) {
  __shared__ bf16 Bs[2 * 128 * 32];
  int lane = threadIdx.x & 31;
  int wave = threadIdx.x >> 5;
  int row  = (blockIdx.x * 4 + wave) * 16;
  int h    = blockIdx.y;
  int n = lane & 15, hi = lane >> 4;

  v8f acc[8] = {};
  gemm_rowblock_lds(Q + h * HDD, DIMD, P, HDD, Bs, row, HDD, acc);

#pragma unroll
  for (int j = 0; j < 8; ++j) {
    int tok = row + j + 8 * hi;
    float mx = -3.4e38f;
#pragma unroll
    for (int t = 0; t < 8; ++t) mx = fmaxf(mx, acc[t][j]);
    mx = hmax16(mx);
#pragma unroll
    for (int t = 0; t < 8; ++t) {
      // -0.5*||x||^2 is constant per row -> cancels under (xp - max)
      float v = __expf((acc[t][j] - mx) * CSCALE) * MINV + EPS_F;
      qphi[(size_t)tok * (HH * MM) + h * MM + 16 * t + n] = (bf16)v;
    }
  }
}

// FAVOR k stage 1: projection GEMM + bias, f32 transposed xp_kT[b][hk][m][s]
__global__ void __launch_bounds__(128)
k_favor_k(const bf16* __restrict__ Kr, const bf16* __restrict__ P,
          const float* __restrict__ ssk, float* __restrict__ xpkT) {
  __shared__ bf16 Bs[2 * 128 * 32];
  int lane = threadIdx.x & 31;
  int wave = threadIdx.x >> 5;
  int row  = (blockIdx.x * 4 + wave) * 16;
  int hk   = blockIdx.y;
  int n = lane & 15, hi = lane >> 4;

  v8f acc[8] = {};
  gemm_rowblock_lds(Kr + hk * HDD, HKK * HDD, P, HDD, Bs, row, HDD, acc);

#pragma unroll
  for (int j = 0; j < 8; ++j) {
    int tok = row + j + 8 * hi;
    int b = tok >> 12, s = tok & (SS - 1);
    float bias = -0.5f * CSCALE * CSCALE * ssk[(size_t)tok * HKK + hk];
#pragma unroll
    for (int t = 0; t < 8; ++t) {
      int m = 16 * t + n;
      xpkT[(((size_t)(b * HKK + hk) * MM + m) * SS) + s] = CSCALE * acc[t][j] + bias;
    }
  }
}

// FAVOR k stage 2: per (b,hk,m) row over S: max, exp, mask, bf16 store + k_sum
__global__ void __launch_bounds__(256)
k_kphi(const float* __restrict__ xp, const int* __restrict__ mask,
       bf16* __restrict__ kphiT, float* __restrict__ ksum) {
  __shared__ float red[256];
  int r = blockIdx.x;              // (b*HK + hk)*128 + m
  int b = r >> 10;
  int tid = threadIdx.x;
  const float* src = xp + (size_t)r * SS;
  const int* mrow = mask + (size_t)b * SS;

  float mx = -3.4e38f;
  for (int s = tid; s < SS; s += 256) mx = fmaxf(mx, src[s]);
  red[tid] = mx; __syncthreads();
  for (int st = 128; st > 0; st >>= 1) {
    if (tid < st) red[tid] = fmaxf(red[tid], red[tid + st]);
    __syncthreads();
  }
  mx = red[0];
  __syncthreads();

  float sum = 0.f;
  for (int s = tid; s < SS; s += 256) {
    float v = __expf(src[s] - mx) * MINV + EPS_F;
    v *= (mrow[s] > 0) ? 1.f : 0.f;
    kphiT[(size_t)r * SS + s] = (bf16)v;
    sum += v;
  }
  red[tid] = sum; __syncthreads();
  for (int st = 128; st > 0; st >>= 1) {
    if (tid < st) red[tid] += red[tid + st];
    __syncthreads();
  }
  if (tid == 0) ksum[r] = red[0];
}

// kv GEMM per (b,hk): kvT[d][m] = sum_s vT[d][s] * kphiT[m][s]. grid=(2, B*HK)
__global__ void __launch_bounds__(128)
k_kv(const bf16* __restrict__ vT, const bf16* __restrict__ kphiT,
     bf16* __restrict__ kvT) {
  __shared__ bf16 Bs[2 * 128 * 32];
  int lane = threadIdx.x & 31;
  int wave = threadIdx.x >> 5;
  int row  = (blockIdx.x * 4 + wave) * 16;   // d-row block
  int head = blockIdx.y;                     // b*HK + hk
  int n = lane & 15, hi = lane >> 4;

  v8f acc[8] = {};
  gemm_rowblock_lds(vT + (size_t)head * HDD * SS, SS,
                    kphiT + (size_t)head * MM * SS, SS, Bs, row, SS, acc);

#pragma unroll
  for (int j = 0; j < 8; ++j) {
    int d = row + j + 8 * hi;
#pragma unroll
    for (int t = 0; t < 8; ++t)
      kvT[((size_t)head * HDD + d) * MM + 16 * t + n] = (bf16)acc[t][j];
  }
}

// denom[tok][h] = sum_m q_phi[tok][h][m] * k_sum[b][hk][m]
__global__ void __launch_bounds__(256)
k_denom(const bf16* __restrict__ qphi, const float* __restrict__ ksum,
        float* __restrict__ denom) {
  int idx = blockIdx.x * 256 + threadIdx.x;  // tok*16 + h
  int tok = idx >> 4, h = idx & 15;
  int b = tok >> 12, hk = h >> 1;
  const bf16* q = qphi + (size_t)tok * (HH * MM) + h * MM;
  const float* ks = ksum + (size_t)(b * HKK + hk) * MM;
  float s = 0.f;
#pragma unroll 8
  for (int m = 0; m < MM; ++m) s += (float)q[m] * ks[m];
  denom[idx] = s;
}

// attention output GEMM per head + normalize + mask -> bf16 attn[tok][h*128+d]
__global__ void __launch_bounds__(128)
k_attn(const bf16* __restrict__ qphi, const bf16* __restrict__ kvT,
       const float* __restrict__ denom, const int* __restrict__ mask,
       bf16* __restrict__ attn) {
  __shared__ bf16 Bs[2 * 128 * 32];
  int lane = threadIdx.x & 31;
  int wave = threadIdx.x >> 5;
  int row  = (blockIdx.x * 4 + wave) * 16;
  int h    = blockIdx.y;
  int n = lane & 15, hi = lane >> 4;
  int b = row >> 12, hk = h >> 1;

  v8f acc[8] = {};
  gemm_rowblock_lds(qphi + h * MM, HH * MM,
                    kvT + (size_t)(b * HKK + hk) * HDD * MM, MM, Bs, row, MM, acc);

#pragma unroll
  for (int j = 0; j < 8; ++j) {
    int tok = row + j + 8 * hi;
    float inv = 1.f / (denom[(size_t)tok * HH + h] + EPS_F);
    float vm = (mask[tok] > 0) ? 1.f : 0.f;
#pragma unroll
    for (int t = 0; t < 8; ++t)
      attn[(size_t)tok * DIMD + h * HDD + 16 * t + n] =
          (bf16)(acc[t][j] * inv * vm);
  }
}

// final output GEMM: y[tok][d] = sum_o attn[tok][o] * wo[d][o]
__global__ void __launch_bounds__(128)
k_out(const bf16* __restrict__ attn, const bf16* __restrict__ wo,
      float* __restrict__ y) {
  __shared__ bf16 Bs[2 * 128 * 32];
  int lane = threadIdx.x & 31;
  int wave = threadIdx.x >> 5;
  int row  = (blockIdx.x * 4 + wave) * 16;
  int colBase = blockIdx.y * 128;
  int n = lane & 15, hi = lane >> 4;

  v8f acc[8] = {};
  gemm_rowblock_lds(attn, DIMD, wo + (size_t)colBase * DIMD, DIMD, Bs, row, DIMD, acc);

#pragma unroll
  for (int j = 0; j < 8; ++j) {
    int tok = row + j + 8 * hi;
#pragma unroll
    for (int t = 0; t < 8; ++t)
      y[(size_t)tok * DIMD + colBase + 16 * t + n] = acc[t][j];
  }
}

// ------------------------- launcher ----------------------------------------
extern "C" void kernel_launch(void* const* d_in, const int* in_sizes, int n_in,
                              void* d_out, int out_size, void* d_ws, size_t ws_size,
                              hipStream_t stream) {
  const float* x    = (const float*)d_in[0];
  const int*   msk  = (const int*)d_in[1];
  const float* fc   = (const float*)d_in[2];
  const float* wq   = (const float*)d_in[3];
  const float* wk   = (const float*)d_in[4];
  const float* wv   = (const float*)d_in[5];
  const float* wo   = (const float*)d_in[6];
  const float* gq   = (const float*)d_in[7];
  const float* gk   = (const float*)d_in[8];
  const float* proj = (const float*)d_in[9];
  float* y = (float*)d_out;

  char* ws = (char*)d_ws;
  size_t off = 0;
  auto alloc = [&](size_t bytes) {
    size_t o = off;
    off += (bytes + 255) & ~(size_t)255;
    return o;
  };

  size_t o_xbf   = alloc((size_t)NTOK * DIMD * 2);          // Xbf ; later xp_kT
  size_t o_wq    = alloc((size_t)DIMD * DIMD * 2);
  size_t o_wk    = alloc((size_t)HKK * HDD * DIMD * 2);
  size_t o_wv    = alloc((size_t)HKK * HDD * DIMD * 2);
  size_t o_wo    = alloc((size_t)DIMD * DIMD * 2);
  size_t o_proj  = alloc((size_t)MM * HDD * 2);
  size_t o_ssk   = alloc((size_t)NTOK * HKK * 4);
  size_t o_ksum  = alloc((size_t)BB * HKK * MM * 4);
  size_t o_denom = alloc((size_t)NTOK * HH * 4);
  size_t o_kvt   = alloc((size_t)BB * HKK * HDD * MM * 2);
  size_t o_qrope = alloc((size_t)NTOK * DIMD * 2);          // q_roped ; later attn
  size_t o_krope = alloc((size_t)NTOK * HKK * HDD * 2);     // k_roped ; later k_phiT
  size_t o_vt    = alloc((size_t)BB * HKK * HDD * SS * 2);
  size_t o_qphi  = alloc((size_t)NTOK * HH * MM * 2);
  (void)ws_size; (void)n_in; (void)in_sizes; (void)out_size;

  bf16*  Xbf    = (bf16*)(ws + o_xbf);
  bf16*  Wq     = (bf16*)(ws + o_wq);
  bf16*  Wk     = (bf16*)(ws + o_wk);
  bf16*  Wv     = (bf16*)(ws + o_wv);
  bf16*  Wo     = (bf16*)(ws + o_wo);
  bf16*  Pb     = (bf16*)(ws + o_proj);
  float* Ssk    = (float*)(ws + o_ssk);
  float* Ksum   = (float*)(ws + o_ksum);
  float* Denom  = (float*)(ws + o_denom);
  bf16*  KvT    = (bf16*)(ws + o_kvt);
  bf16*  Qrope  = (bf16*)(ws + o_qrope);
  bf16*  Krope  = (bf16*)(ws + o_krope);
  bf16*  VT     = (bf16*)(ws + o_vt);
  bf16*  Qphi   = (bf16*)(ws + o_qphi);
  // aliased regions (disjoint lifetimes)
  float* XpkT   = (float*)(ws + o_xbf);    // after Xbf is dead
  bf16*  KphiT  = (bf16*)(ws + o_krope);   // after Krope is dead
  bf16*  Attn   = (bf16*)(ws + o_qrope);   // after Qrope is dead

  // 0) convert to bf16
  k_cvt<<<2048, 256, 0, stream>>>(x,    Xbf, NTOK * DIMD);
  k_cvt<<<1024, 256, 0, stream>>>(wq,   Wq,  DIMD * DIMD);
  k_cvt<<<512,  256, 0, stream>>>(wk,   Wk,  HKK * HDD * DIMD);
  k_cvt<<<512,  256, 0, stream>>>(wv,   Wv,  HKK * HDD * DIMD);
  k_cvt<<<1024, 256, 0, stream>>>(wo,   Wo,  DIMD * DIMD);
  k_cvt<<<16,   256, 0, stream>>>(proj, Pb,  MM * HDD);

  // 1) QKV projections (+ fused RMSNorm/RoPE for Q,K; transposed store for V)
  k_qk_proj<<<dim3(NTOK / 64, HH),  128, 0, stream>>>(Xbf, Wq, gq, fc, Qrope, nullptr, HH * HDD);
  k_qk_proj<<<dim3(NTOK / 64, HKK), 128, 0, stream>>>(Xbf, Wk, gk, fc, Krope, Ssk,     HKK * HDD);
  k_v_proj <<<dim3(NTOK / 64, HKK), 128, 0, stream>>>(Xbf, Wv, VT);

  // 2) FAVOR features
  k_favor_q<<<dim3(NTOK / 64, HH),  128, 0, stream>>>(Qrope, Pb, Qphi);
  k_favor_k<<<dim3(NTOK / 64, HKK), 128, 0, stream>>>(Krope, Pb, Ssk, XpkT);
  k_kphi   <<<BB * HKK * MM, 256, 0, stream>>>(XpkT, msk, KphiT, Ksum);

  // 3) linear attention
  k_kv   <<<dim3(HDD / 64, BB * HKK), 128, 0, stream>>>(VT, KphiT, KvT);
  k_denom<<<NTOK * HH / 256, 256, 0, stream>>>(Qphi, Ksum, Denom);
  k_attn <<<dim3(NTOK / 64, HH), 128, 0, stream>>>(Qphi, KvT, Denom, msk, Attn);

  // 4) output projection
  k_out<<<dim3(NTOK / 64, DIMD / 128), 128, 0, stream>>>(Attn, Wo, y);
}
